// GraphAttentionNetwork_22625887715607
// MI455X (gfx1250) — compile-verified, matrix operand order checked by reading, not run on hardware
//
#include <hip/hip_runtime.h>
#include <math.h>
#include <type_traits>

typedef __attribute__((ext_vector_type(16))) _Float16 v16h;
typedef __attribute__((ext_vector_type(8)))  _Float16 v8h;
typedef __attribute__((ext_vector_type(8)))  float    v8f;

#define TPB 256
#define GAT_NEG_SLOPE 0.2f
#define BS_STRIDE 40   // f16 elements per LDS B-tile row (padded vs 32 to spread banks)

// ---------------- utility device helpers ----------------

static __device__ __forceinline__ void atomicMaxF(float* addr, float val) {
  if (val >= 0.0f) atomicMax((int*)addr, __float_as_int(val));
  else             atomicMin((unsigned int*)addr, __float_as_uint(val));
}

static __device__ __forceinline__ void edge_sd(const int* __restrict__ ei, int E,
                                               int e, int& s, int& d) {
  if (e < E) { s = ei[e]; d = ei[E + e]; }
  else       { s = e - E; d = e - E; }       // appended self-loops
}

// ---------------- conversion / fill kernels ----------------

__global__ void k_f32_to_f16(const float* __restrict__ in, _Float16* __restrict__ out, int n) {
  int i = blockIdx.x * blockDim.x + threadIdx.x;
  if (i < n) out[i] = (_Float16)in[i];
}

// Wt[n][k] = (half) W[k][n]   (W is K x Nout row-major)
__global__ void k_w_transpose_f16(const float* __restrict__ W, _Float16* __restrict__ Wt,
                                  int K, int Nout) {
  int i = blockIdx.x * blockDim.x + threadIdx.x;
  if (i < K * Nout) {
    int k = i / Nout, nn = i - k * Nout;
    Wt[(size_t)nn * K + k] = (_Float16)W[i];
  }
}

__global__ void k_fill(float* __restrict__ p, float v, int n) {
  int i = blockIdx.x * blockDim.x + threadIdx.x;
  if (i < n) p[i] = v;
}

// ---------------- WMMA GEMM ----------------
// C[M,N] = A[M,K] * B[K,N].  A: f16 row-major; Bt: f16 row-major [N x K] (B^T).
// Block = 8 waves: 128 M-rows x 64 N-cols. B tile (64x32) staged in LDS (double
// buffered, one barrier per K tile); A fragment software-pipelined in registers.
// Epilogue writes C in f32 (for logits) and f16 (for the edge gather).

static __device__ __forceinline__ v16h load_frag(const _Float16* __restrict__ p) {
  v8h lo = *(const v8h*)(p);
  v8h hi = *(const v8h*)(p + 16);
  v16h r;
#pragma unroll
  for (int i = 0; i < 8; ++i) { r[i] = lo[i]; r[i + 8] = hi[i]; }
  return r;
}

__global__ void __launch_bounds__(TPB)
k_wmma_gemm(const _Float16* __restrict__ A, const _Float16* __restrict__ Bt,
            float* __restrict__ C, _Float16* __restrict__ Ch, int M, int K, int Nn) {
  __shared__ _Float16 Bs[2][64 * BS_STRIDE];   // ~10 KB

  const int tid    = threadIdx.x;
  const int lane   = tid & 31;
  const int wid    = __builtin_amdgcn_readfirstlane(tid >> 5);  // SGPR -> scalar branches
  const int tilesM = M >> 4;
  const int blocksM = (tilesM + 7) >> 3;
  const int bm = (int)(blockIdx.x % blocksM);
  const int tg = (int)(blockIdx.x / blocksM);
  const int tm = bm * 8 + wid;                 // this wave's M tile (scalar)
  const bool active = (tm < tilesM);           // scalar condition -> s_cbranch
  const int lo = lane & 15;
  const int hi = lane >> 4;
  const int T  = K >> 5;                       // number of 32-wide K tiles

  // cooperative B-tile staging: each thread moves 8 f16 (16 B)
  const int ln = tid >> 2;                     // 0..63 : n within tile
  const int lk = (tid & 3) * 8;                // 0,8,16,24 : k within tile
  const _Float16* bsrc = Bt + (size_t)(tg * 64 + ln) * K + lk;

  const _Float16* ap = A + (size_t)((active ? tm : 0) * 16 + lo) * K + hi * 8;

  v8f acc0 = {}, acc1 = {}, acc2 = {}, acc3 = {};

  // prologue: stage B tile 0 in LDS, A tile 0 in registers
  {
    v8h b0 = *(const v8h*)(bsrc);
    *(v8h*)(&Bs[0][ln * BS_STRIDE + lk]) = b0;
  }
  v16h a = {};
  if (active) a = load_frag(ap);
  __syncthreads();

  for (int t = 0; t < T; ++t) {
    const int p = t & 1;
    // issue next-tile global loads first: they fly under the WMMAs below
    v8h bnext = {};
    if (t + 1 < T) bnext = *(const v8h*)(bsrc + (size_t)(t + 1) * 32);
    v16h anext = {};
    if (active && (t + 1 < T)) {
      const _Float16* an = ap + (t + 1) * 32;
      __builtin_prefetch(an + 32, 0, 3);
      anext = load_frag(an);
    }

    if (active) {
      v16h b0 = load_frag(&Bs[p][( 0 + lo) * BS_STRIDE + hi * 8]);
      v16h b1 = load_frag(&Bs[p][(16 + lo) * BS_STRIDE + hi * 8]);
      v16h b2 = load_frag(&Bs[p][(32 + lo) * BS_STRIDE + hi * 8]);
      v16h b3 = load_frag(&Bs[p][(48 + lo) * BS_STRIDE + hi * 8]);
      acc0 = __builtin_amdgcn_wmma_f32_16x16x32_f16(false, a, false, b0, (short)0, acc0, false, false);
      acc1 = __builtin_amdgcn_wmma_f32_16x16x32_f16(false, a, false, b1, (short)0, acc1, false, false);
      acc2 = __builtin_amdgcn_wmma_f32_16x16x32_f16(false, a, false, b2, (short)0, acc2, false, false);
      acc3 = __builtin_amdgcn_wmma_f32_16x16x32_f16(false, a, false, b3, (short)0, acc3, false, false);
    }
    if (t + 1 < T) {
      *(v8h*)(&Bs[p ^ 1][ln * BS_STRIDE + lk]) = bnext;  // other buffer: no reader conflict
      __syncthreads();
    }
    a = anext;
  }

  if (!active) return;
  // C/D layout: VGPR r -> row (r + 8*hi), col = lo
  const size_t rbase = (size_t)(tm * 16 + hi * 8) * Nn + tg * 64 + lo;
  float*    crow = C  + rbase;
  _Float16* hrow = Ch + rbase;
#pragma unroll
  for (int r = 0; r < 8; ++r) {
    crow[(size_t)r * Nn +  0] = acc0[r];
    crow[(size_t)r * Nn + 16] = acc1[r];
    crow[(size_t)r * Nn + 32] = acc2[r];
    crow[(size_t)r * Nn + 48] = acc3[r];
    hrow[(size_t)r * Nn +  0] = (_Float16)acc0[r];
    hrow[(size_t)r * Nn + 16] = (_Float16)acc1[r];
    hrow[(size_t)r * Nn + 32] = (_Float16)acc2[r];
    hrow[(size_t)r * Nn + 48] = (_Float16)acc3[r];
  }
}

// ---------------- attention logits:  AS[n,h] = <h[n,h,:], a_src[h,:]> ----------------

template <int H, int C>
__global__ void k_logits(const float* __restrict__ Hf, const float* __restrict__ a_src,
                         const float* __restrict__ a_dst, float* __restrict__ AS,
                         float* __restrict__ AD, int n) {
  int i = blockIdx.x * blockDim.x + threadIdx.x;
  if (i >= n * H) return;
  int node = i / H, h = i & (H - 1);
  const float* hp = Hf + (size_t)node * H * C + (size_t)h * C;
  const float* as = a_src + (size_t)h * C;
  const float* ad = a_dst + (size_t)h * C;
  float s1 = 0.f, s2 = 0.f;
#pragma unroll 8
  for (int c = 0; c < C; ++c) { float v = hp[c]; s1 += v * as[c]; s2 += v * ad[c]; }
  AS[i] = s1; AD[i] = s2;
}

// ---------------- segment softmax, 3 passes over edges ----------------

template <int H>
__global__ void k_edge_max(const int* __restrict__ ei, int E, int Etot,
                           const float* __restrict__ AS, const float* __restrict__ AD,
                           float* __restrict__ EV, float* __restrict__ Mx) {
  int i = blockIdx.x * blockDim.x + threadIdx.x;
  if (i >= Etot * H) return;
  int e = i / H, h = i & (H - 1);
  int s, d; edge_sd(ei, E, e, s, d);
  float v = AS[s * H + h] + AD[d * H + h];
  v = v > 0.f ? v : GAT_NEG_SLOPE * v;      // leaky_relu
  EV[i] = v;
  atomicMaxF(&Mx[d * H + h], v);
}

template <int H>
__global__ void k_edge_exp(const int* __restrict__ ei, int E, int Etot,
                           float* __restrict__ EV, const float* __restrict__ Mx,
                           float* __restrict__ Ss) {
  int i = blockIdx.x * blockDim.x + threadIdx.x;
  if (i >= Etot * H) return;
  int e = i / H, h = i & (H - 1);
  int s, d; edge_sd(ei, E, e, s, d);
  float v = __expf(EV[i] - Mx[d * H + h]);
  EV[i] = v;
  atomicAdd(&Ss[d * H + h], v);
}

// gather f16 features (halves random-read traffic), fp32 atomic accumulate
template <int H, int C>
__global__ void k_edge_agg(const int* __restrict__ ei, int E, int Etot,
                           const float* __restrict__ EV, const float* __restrict__ Ss,
                           const _Float16* __restrict__ Hh, float* __restrict__ Agg) {
  constexpr int HC = H * C;               // power of two (512 or 64)
  int i = blockIdx.x * blockDim.x + threadIdx.x;  // over Etot*HC (<2^31 here)
  if (i >= Etot * HC) return;
  int e = i / HC, j = i & (HC - 1);
  int h = j / C;                          // constant divisor -> shift
  int s, d; edge_sd(ei, E, e, s, d);
  float alpha = EV[e * H + h] / (Ss[d * H + h] + 1e-16f);
  float msg = (float)Hh[(size_t)s * HC + j] * alpha;
  atomicAdd(&Agg[(size_t)d * HC + j], msg);
}

// ---------------- bias + activation ----------------

__global__ void k_bias_elu(float* __restrict__ Agg, const float* __restrict__ b, int n, int F) {
  int i = blockIdx.x * blockDim.x + threadIdx.x;
  if (i >= n * F) return;
  float v = Agg[i] + b[i & (F - 1)];      // F is a power of two (512)
  Agg[i] = v > 0.f ? v : (__expf(v) - 1.f);   // elu
}

__global__ void k_bias_out(const float* __restrict__ Agg, const float* __restrict__ b,
                           float* __restrict__ out, int n, int F) {
  int i = blockIdx.x * blockDim.x + threadIdx.x;
  if (i >= n * F) return;
  out[i] = Agg[i] + b[i & (F - 1)];       // F is a power of two (64)
}

// ---------------- host orchestration ----------------

extern "C" void kernel_launch(void* const* d_in, const int* in_sizes, int n_in,
                              void* d_out, int out_size, void* d_ws, size_t ws_size,
                              hipStream_t stream) {
  const float* x   = (const float*)d_in[0];
  const int*   ei  = (const int*)  d_in[1];
  const float* W1  = (const float*)d_in[2];
  const float* as1 = (const float*)d_in[3];
  const float* ad1 = (const float*)d_in[4];
  const float* b1  = (const float*)d_in[5];
  const float* W2  = (const float*)d_in[6];
  const float* as2 = (const float*)d_in[7];
  const float* ad2 = (const float*)d_in[8];
  const float* b2  = (const float*)d_in[9];
  const float* W3  = (const float*)d_in[10];
  const float* as3 = (const float*)d_in[11];
  const float* ad3 = (const float*)d_in[12];
  const float* b3  = (const float*)d_in[13];
  (void)n_in; (void)out_size; (void)ws_size;

  const int DIN = 128, HID = 128, HEADS = 4, DOUT = 64;
  const int n    = in_sizes[0] / DIN;   // 50000 (multiple of 16)
  const int E    = in_sizes[1] / 2;     // 400000
  const int Etot = E + n;               // + self loops
  const int F    = HEADS * HID;         // 512

  // carve workspace
  char* w = (char*)d_ws;
  auto carve = [&](size_t bytes) -> void* {
    void* p = (void*)w; w += (bytes + 255) & ~(size_t)255; return p;
  };
  _Float16* A16  = (_Float16*)carve((size_t)n * F * sizeof(_Float16));
  _Float16* Wt   = (_Float16*)carve((size_t)F * F * sizeof(_Float16));
  _Float16* Hf16 = (_Float16*)carve((size_t)n * F * sizeof(_Float16));
  float* Hf  = (float*)carve((size_t)n * F * sizeof(float));
  float* AS  = (float*)carve((size_t)n * HEADS * sizeof(float));
  float* AD  = (float*)carve((size_t)n * HEADS * sizeof(float));
  float* Mx  = (float*)carve((size_t)n * HEADS * sizeof(float));
  float* Ss  = (float*)carve((size_t)n * HEADS * sizeof(float));
  float* EV  = (float*)carve((size_t)Etot * HEADS * sizeof(float));
  float* Agg = (float*)carve((size_t)n * F * sizeof(float));

  auto cdiv = [](long long a, long long b) { return (unsigned)((a + b - 1) / b); };

  auto run_layer = [&](const float* act, int K, int Nout,
                       const float* Wp, const float* asp, const float* adp,
                       auto HTag, auto CTag) {
    constexpr int H = decltype(HTag)::value;
    constexpr int C = decltype(CTag)::value;
    k_f32_to_f16<<<cdiv((long long)n * K, TPB), TPB, 0, stream>>>(act, A16, n * K);
    k_w_transpose_f16<<<cdiv((long long)K * Nout, TPB), TPB, 0, stream>>>(Wp, Wt, K, Nout);
    long long blocksM = ((n / 16) + 7) / 8;
    long long grid = blocksM * (Nout / 64);
    k_wmma_gemm<<<(unsigned)grid, TPB, 0, stream>>>(A16, Wt, Hf, Hf16, n, K, Nout);
    k_logits<H, C><<<cdiv((long long)n * H, TPB), TPB, 0, stream>>>(Hf, asp, adp, AS, AD, n);
    k_fill<<<cdiv((long long)n * H, TPB), TPB, 0, stream>>>(Mx, -INFINITY, n * H);
    k_fill<<<cdiv((long long)n * H, TPB), TPB, 0, stream>>>(Ss, 0.f, n * H);
    k_fill<<<cdiv((long long)n * Nout, TPB), TPB, 0, stream>>>(Agg, 0.f, n * Nout);
    k_edge_max<H><<<cdiv((long long)Etot * H, TPB), TPB, 0, stream>>>(ei, E, Etot, AS, AD, EV, Mx);
    k_edge_exp<H><<<cdiv((long long)Etot * H, TPB), TPB, 0, stream>>>(ei, E, Etot, EV, Mx, Ss);
    k_edge_agg<H, C><<<cdiv((long long)Etot * H * C, TPB), TPB, 0, stream>>>(ei, E, Etot, EV, Ss, Hf16, Agg);
  };

  // layer 1: DIN -> 4x128 (concat), elu
  run_layer(x, DIN, F, W1, as1, ad1,
            std::integral_constant<int, 4>{}, std::integral_constant<int, 128>{});
  k_bias_elu<<<cdiv((long long)n * F, TPB), TPB, 0, stream>>>(Agg, b1, n, F);
  // layer 2: 512 -> 4x128 (concat), elu   (Agg is read into f16 before being re-zeroed)
  run_layer(Agg, F, F, W2, as2, ad2,
            std::integral_constant<int, 4>{}, std::integral_constant<int, 128>{});
  k_bias_elu<<<cdiv((long long)n * F, TPB), TPB, 0, stream>>>(Agg, b2, n, F);
  // layer 3: 512 -> 64, heads=1
  run_layer(Agg, F, DOUT, W3, as3, ad3,
            std::integral_constant<int, 1>{}, std::integral_constant<int, 64>{});
  k_bias_out<<<cdiv((long long)n * DOUT, TPB), TPB, 0, stream>>>(Agg, b3, (float*)d_out, n, DOUT);
}
